// AttentionDecoder_85452669321783
// MI455X (gfx1250) — matrix operand order repaired and checked by reading, not compiled, not run on hardware
//
#include <hip/hip_runtime.h>

// ---------------- problem constants (from reference) ----------------
namespace cfg {
constexpr int B    = 128;
constexpr int P    = 196;
constexpr int ENC  = 2048;
constexpr int A    = 512;
constexpr int E    = 512;
constexpr int H    = 512;
constexpr int V    = 20000;
constexpr int TMAX = 20;
}

// ---------------- types / bf16 helpers ----------------
typedef __attribute__((ext_vector_type(16))) __bf16         v16bf;
typedef __attribute__((ext_vector_type(8)))  float          v8f;
typedef __attribute__((ext_vector_type(16))) unsigned short v16u;
typedef __attribute__((ext_vector_type(8)))  unsigned short v8u;
typedef __attribute__((ext_vector_type(4)))  float          f4;

__device__ __forceinline__ unsigned short f2bfu(float f) {
  unsigned u = __float_as_uint(f);
  u += 0x7FFFu + ((u >> 16) & 1u);           // round-to-nearest-even
  return (unsigned short)(u >> 16);
}
__device__ __forceinline__ float bfu2f(unsigned short h) {
  return __uint_as_float((unsigned)h << 16);
}

// ---------------- bf16 WMMA GEMM: C = A(f32) * B(f32) + bias ---------
// A: [M,K] row-major (lda). B: !BT -> [K,N] (ldb); BT -> [N,K] (ldb).
// LDS tiles are stored in WMMA fragment-native layout so every lane reads
// its whole 16x bf16 fragment as one contiguous 32B LDS access. All B
// fragments are preloaded into registers so the 8 WMMAs issue back-to-back
// behind partial dscnt waits instead of one full LDS round-trip each.
template <bool BT>
__global__ __launch_bounds__(128)
void gemm_bf16_wmma(const float* __restrict__ Ag, int lda,
                    const float* __restrict__ Bg, int ldb,
                    const float* __restrict__ bias,
                    float* __restrict__ Cg, long long ldc,
                    int M, int N, int K) {
  constexpr int BM = 64, BN = 128, BK = 32;
  // fragment-native LDS: [frag][lane][16 halfwords]
  __shared__ alignas(32) unsigned short As_l[4 * 32 * 16];   // 4 M-frags
  __shared__ alignas(32) unsigned short Bs_l[8 * 32 * 16];   // 8 N-frags

  const int tid  = threadIdx.x;
  const int lane = tid & 31;
  const int wave = tid >> 5;      // 0..3 -> 16-row M strip
  const int half = lane >> 4;
  const int r    = lane & 15;
  const int m0   = blockIdx.x * BM;
  const int n0   = blockIdx.y * BN;

  v8f acc[8];
  #pragma unroll
  for (int i = 0; i < 8; ++i)
    #pragma unroll
    for (int j = 0; j < 8; ++j) acc[i][j] = 0.f;

  for (int k0 = 0; k0 < K; k0 += BK) {
    // ---- A tile 64x32: one K-octet (8 f32 -> 8 bf16 -> 1x ds_store_b128) ----
    for (int it = tid; it < 64 * 4; it += 128) {
      const int row = it >> 2;
      const int k8  = (it & 3) * 8;
      const int gm  = m0 + row;
      f4 v0, v1;
      if (gm < M) {
        const float* src = Ag + (long long)gm * lda + (k0 + k8);
        v0 = *(const f4*)(src);
        v1 = *(const f4*)(src + 4);
        if (k0 + BK < K) __builtin_prefetch(src + BK, 0, 0);
      } else {
        v0 = f4{0.f, 0.f, 0.f, 0.f}; v1 = f4{0.f, 0.f, 0.f, 0.f};
      }
      v8u u;
      u[0] = f2bfu(v0.x); u[1] = f2bfu(v0.y); u[2] = f2bfu(v0.z); u[3] = f2bfu(v0.w);
      u[4] = f2bfu(v1.x); u[5] = f2bfu(v1.y); u[6] = f2bfu(v1.z); u[7] = f2bfu(v1.w);
      const int frag  = row >> 4;
      const int hh    = (k8 >> 3) & 1;        // A layout: half from k8
      const int ibase = (k8 & 16) >> 1;       // 0 or 8
      const int ln    = hh * 16 + (row & 15);
      *(v8u*)&As_l[(frag * 32 + ln) * 16 + ibase] = u;
    }
    // ---- B tile 32x128: one K-octet along k for fixed n ----
    for (int it = tid; it < 128 * 4; it += 128) {
      const int n  = it >> 2;
      const int k8 = (it & 3) * 8;
      const int gn = n0 + n;
      v8u u;
      if (gn < N) {
        if (BT) {
          const float* src = Bg + (long long)gn * ldb + (k0 + k8);
          f4 v0 = *(const f4*)(src);
          f4 v1 = *(const f4*)(src + 4);
          if (k0 + BK < K) __builtin_prefetch(src + BK, 0, 0);
          u[0] = f2bfu(v0.x); u[1] = f2bfu(v0.y); u[2] = f2bfu(v0.z); u[3] = f2bfu(v0.w);
          u[4] = f2bfu(v1.x); u[5] = f2bfu(v1.y); u[6] = f2bfu(v1.z); u[7] = f2bfu(v1.w);
        } else {
          #pragma unroll
          for (int j = 0; j < 8; ++j)
            u[j] = f2bfu(Bg[(long long)(k0 + k8 + j) * ldb + gn]);
        }
      } else {
        #pragma unroll
        for (int j = 0; j < 8; ++j) u[j] = 0;
      }
      const int ns    = n >> 4;
      const int hh    = (k8 >> 3) & 1;        // B layout: half from k8
      const int ibase = (k8 & 16) >> 1;       // 0 or 8
      const int ln    = hh * 16 + (n & 15);
      *(v8u*)&Bs_l[(ns * 32 + ln) * 16 + ibase] = u;
    }
    __syncthreads();

    // ---- preload A + all 8 B fragments (32B contiguous LDS reads each),
    //      then fire the 8 WMMAs back-to-back ----
    v16bf afrag = __builtin_bit_cast(
        v16bf, *(const v16u*)&As_l[(wave * 32 + lane) * 16]);
    v16bf bfrag[8];
    #pragma unroll
    for (int ns = 0; ns < 8; ++ns)
      bfrag[ns] = __builtin_bit_cast(
          v16bf, *(const v16u*)&Bs_l[(ns * 32 + lane) * 16]);
    #pragma unroll
    for (int ns = 0; ns < 8; ++ns)
      acc[ns] = __builtin_amdgcn_wmma_f32_16x16x32_bf16(
          false, afrag, false, bfrag[ns], (short)0, acc[ns], false, false);
    __syncthreads();
  }

  // ---- store (ISA f32 C/D layout: M = half*8 + j, N = r) ----
  #pragma unroll
  for (int ns = 0; ns < 8; ++ns) {
    int gn = n0 + ns * 16 + r;
    if (gn < N) {
      float bv = bias ? bias[gn] : 0.f;
      #pragma unroll
      for (int j = 0; j < 8; ++j) {
        int gm = m0 + wave * 16 + half * 8 + j;
        if (gm < M) Cg[(long long)gm * ldc + gn] = acc[ns][j] + bv;
      }
    }
  }
}

// ---------------- img f32 -> bf16 (keeps img L2-resident: 103MB < 192MB) ----
__global__ void cvt_f32_bf16(const float* __restrict__ in,
                             unsigned short* __restrict__ out, long long n) {
  long long i = (long long)blockIdx.x * blockDim.x + threadIdx.x;
  if (i < n) out[i] = f2bfu(in[i]);
}

// ---------------- init: outputs at t=0, caplen, h=0 ----------------
__global__ void init_outputs(float* __restrict__ pred, float* __restrict__ alphas,
                             float* __restrict__ caplen_out,
                             const int* __restrict__ caplen,
                             float* __restrict__ h) {
  using namespace cfg;
  long long idx = (long long)blockIdx.x * blockDim.x + threadIdx.x;
  if (idx < (long long)B * V) {
    int b = (int)(idx / V), v = (int)(idx % V);
    pred[(long long)b * TMAX * V + v] = (v == 1) ? 1.f : 0.f;   // predictions[:,0,:]
  }
  if (idx < (long long)B * P) {
    int b = (int)(idx / P), p = (int)(idx % P);
    alphas[(long long)b * TMAX * P + p] = 0.f;                  // alphas[:,0,:]
  }
  if (idx < (long long)B * H) h[idx] = 0.f;
  if (idx < B) caplen_out[idx] = (float)caplen[idx];
}

// ---------------- attention: scores -> softmax -> ctx -> x ----------------
__global__ __launch_bounds__(256)
void attention_step(const float* __restrict__ att1,   // [B,P,A]
                    const float* __restrict__ att2,   // [B,A]
                    const float* __restrict__ Wf,     // [A]
                    const float* __restrict__ bfp,    // [1]
                    const unsigned short* __restrict__ imgb, // [B,P,ENC] bf16
                    const float* __restrict__ emb,    // [V,E]
                    const int* __restrict__ cap,      // [B,TMAX]
                    int t,
                    float* __restrict__ x,            // [B, E+ENC]
                    float* __restrict__ alphas) {     // [B,TMAX,P]
  using namespace cfg;
  __shared__ float att2_s[A];
  __shared__ float wf_s[A];
  __shared__ float sc[P];
  __shared__ float red[256];
  const int b    = blockIdx.x;
  const int tid  = threadIdx.x;
  const int lane = tid & 31;
  const int wave = tid >> 5;

  for (int i = tid; i < A; i += 256) {
    att2_s[i] = att2[b * A + i];
    wf_s[i]   = Wf[i];
  }
  __syncthreads();

  // scores[p] = relu(att1[b,p,:] + att2[b,:]) . Wf + bf   (one wave per pixel)
  const float* att1b = att1 + (size_t)b * P * A;
  for (int p = wave; p < P; p += 8) {
    float sum = 0.f;
    const float* row = att1b + (size_t)p * A;
    for (int a = lane; a < A; a += 32) {
      float v = row[a] + att2_s[a];
      v = v > 0.f ? v : 0.f;
      sum += v * wf_s[a];
    }
    #pragma unroll
    for (int o = 16; o > 0; o >>= 1) sum += __shfl_xor(sum, o, 32);
    if (lane == 0) sc[p] = sum + bfp[0];
  }
  __syncthreads();

  // softmax over P=196
  float sv = (tid < P) ? sc[tid] : -3.4e38f;
  red[tid] = sv; __syncthreads();
  for (int s = 128; s > 0; s >>= 1) {
    if (tid < s) red[tid] = fmaxf(red[tid], red[tid + s]);
    __syncthreads();
  }
  float mx = red[0]; __syncthreads();
  float ev = (tid < P) ? __expf(sv - mx) : 0.f;
  red[tid] = ev; __syncthreads();
  for (int s = 128; s > 0; s >>= 1) {
    if (tid < s) red[tid] += red[tid + s];
    __syncthreads();
  }
  float inv = 1.f / red[0]; __syncthreads();
  if (tid < P) {
    float al = ev * inv;
    sc[tid] = al;
    alphas[((size_t)b * TMAX + t) * P + tid] = al;
  }
  __syncthreads();

  // ctx[b,e] = sum_p alpha[p] * img[b,p,e]  (bf16 img from L2)
  const unsigned short* ib = imgb + (size_t)b * P * ENC;
  for (int e = tid; e < ENC; e += 256) {
    float acc = 0.f;
    for (int p = 0; p < P; ++p)
      acc += sc[p] * bfu2f(ib[(size_t)p * ENC + e]);
    x[(size_t)b * (E + ENC) + E + e] = acc;
  }
  // x[:, :E] = emb[cap[b, t-1]]
  int w = cap[b * TMAX + (t - 1)];
  for (int j = tid; j < E; j += 256)
    x[(size_t)b * (E + ENC) + j] = emb[(size_t)w * E + j];
}

// ---------------- GRU pointwise update ----------------
__global__ void gru_update(const float* __restrict__ gi,
                           const float* __restrict__ gh,
                           float* __restrict__ h) {
  using namespace cfg;
  int idx = blockIdx.x * blockDim.x + threadIdx.x;
  if (idx >= B * H) return;
  int b = idx / H, j = idx % H;
  const float* gib = gi + (size_t)b * 3 * H;
  const float* ghb = gh + (size_t)b * 3 * H;
  float rg = 1.f / (1.f + __expf(-(gib[j]         + ghb[j])));
  float zg = 1.f / (1.f + __expf(-(gib[H + j]     + ghb[H + j])));
  float ng = tanhf(gib[2 * H + j] + rg * ghb[2 * H + j]);
  float hp = h[idx];
  h[idx] = (1.f - zg) * ng + zg * hp;
}

// ---------------- launch ----------------
extern "C" void kernel_launch(void* const* d_in, const int* in_sizes, int n_in,
                              void* d_out, int out_size, void* d_ws, size_t ws_size,
                              hipStream_t stream) {
  using namespace cfg;
  (void)in_sizes; (void)n_in; (void)out_size; (void)ws_size;

  const float* img    = (const float*)d_in[0];
  const int*   cap    = (const int*)  d_in[1];
  const int*   caplen = (const int*)  d_in[2];
  const float* emb    = (const float*)d_in[3];
  const float* We     = (const float*)d_in[4];
  const float* be     = (const float*)d_in[5];
  const float* Wd     = (const float*)d_in[6];
  const float* bd     = (const float*)d_in[7];
  const float* Wf     = (const float*)d_in[8];
  const float* bfp    = (const float*)d_in[9];
  const float* W_ih   = (const float*)d_in[10];
  const float* b_ih   = (const float*)d_in[11];
  const float* W_hh   = (const float*)d_in[12];
  const float* b_hh   = (const float*)d_in[13];
  const float* Wo     = (const float*)d_in[14];
  const float* bo     = (const float*)d_in[15];

  float* out        = (float*)d_out;
  float* pred       = out;                                   // [B,TMAX,V]
  float* caplen_out = out + (size_t)B * TMAX * V;            // [B]
  float* alphas     = caplen_out + B;                        // [B,TMAX,P]

  // workspace carve-out
  char* wsp = (char*)d_ws;
  auto take = [&](size_t bytes) -> void* {
    void* p = (void*)wsp;
    wsp += (bytes + 255) & ~(size_t)255;
    return p;
  };
  const long long nimg = (long long)B * P * ENC;
  unsigned short* img_bf = (unsigned short*)take((size_t)nimg * 2);           // 103 MB
  float* att1 = (float*)take((size_t)B * P * A * 4);                          // 51.4 MB
  float* att2 = (float*)take((size_t)B * A * 4);
  float* h    = (float*)take((size_t)B * H * 4);
  float* x    = (float*)take((size_t)B * (E + ENC) * 4);
  float* gi   = (float*)take((size_t)B * 3 * H * 4);
  float* gh   = (float*)take((size_t)B * 3 * H * 4);

  // one-time: init outputs/h, bf16 img, hoisted att1 GEMM
  {
    long long ninit = (long long)B * V;
    init_outputs<<<(unsigned)((ninit + 255) / 256), 256, 0, stream>>>(
        pred, alphas, caplen_out, caplen, h);
    cvt_f32_bf16<<<(unsigned)((nimg + 255) / 256), 256, 0, stream>>>(img, img_bf, nimg);
    // att1[B*P, A] = img[B*P, ENC] @ We[ENC, A] + be
    gemm_bf16_wmma<false><<<dim3((B * P) / 64, A / 128), 128, 0, stream>>>(
        img, ENC, We, A, be, att1, A, B * P, A, ENC);
  }

  // sequential scan: t = 1 .. TMAX-1 (step t consumes word_emb[:, t-1])
  for (int t = 1; t < TMAX; ++t) {
    // att2 = h @ Wd + bd
    gemm_bf16_wmma<false><<<dim3(B / 64, A / 128), 128, 0, stream>>>(
        h, H, Wd, A, bd, att2, A, B, A, H);
    // softmax attention -> alphas[:,t,:], ctx -> x[:,E:], gather emb -> x[:,:E]
    attention_step<<<B, 256, 0, stream>>>(att1, att2, Wf, bfp, img_bf, emb, cap,
                                          t, x, alphas);
    // gi = x @ W_ih^T + b_ih ; gh = h @ W_hh^T + b_hh
    gemm_bf16_wmma<true><<<dim3(B / 64, (3 * H) / 128), 128, 0, stream>>>(
        x, E + ENC, W_ih, E + ENC, b_ih, gi, 3 * H, B, 3 * H, E + ENC);
    gemm_bf16_wmma<true><<<dim3(B / 64, (3 * H) / 128), 128, 0, stream>>>(
        h, H, W_hh, H, b_hh, gh, 3 * H, B, 3 * H, H);
    // GRU pointwise: h <- (1-z)*n + z*h
    gru_update<<<(B * H + 255) / 256, 256, 0, stream>>>(gi, gh, h);
    // predictions[:, t, :] = h @ Wo^T + bo
    gemm_bf16_wmma<true><<<dim3(B / 64, (V + 127) / 128), 128, 0, stream>>>(
        h, H, Wo, H, bo, pred + (long long)t * V, (long long)TMAX * V, B, V, H);
  }
}
